// GNNRes_35510789603460
// MI455X (gfx1250) — matrix operand order repaired and checked by reading, not compiled, not run on hardware
//
#include <hip/hip_runtime.h>
#include <cstdint>
#include <cstddef>

typedef _Float16 h8   __attribute__((ext_vector_type(8)));
typedef _Float16 v16h __attribute__((ext_vector_type(16)));
typedef float    v8f  __attribute__((ext_vector_type(8)));

__device__ __forceinline__ void atomAddF(float* p, float v) {
  unsafeAtomicAdd(p, v);   // hardware global_atomic_add_f32
}

// ---------------- weight pack: B in WMMA column-major layout ----------------
// dst[n*strideK + koff + kk] = (n<Nsrc && kk<Kvalid) ? (f16)W[(krowoff+kk)*Nld + n] : 0
__global__ void k_pack(const float* __restrict__ W, _Float16* __restrict__ dst,
                       int Kvalid, int Nsrc, int Nld, int Npad, int strideK,
                       int koff, int krowoff) {
  int t = blockIdx.x * blockDim.x + threadIdx.x;
  if (t >= Npad * 128) return;
  int n  = t >> 7;
  int kk = t & 127;
  float v = 0.f;
  if (n < Nsrc && kk < Kvalid) v = W[(size_t)(krowoff + kk) * Nld + n];
  dst[(size_t)n * strideK + koff + kk] = (_Float16)v;
}

// ---------------- f32 [N,C] -> f16 [N,stride] (segment of 128 cols) --------
__global__ void k_conv16(const float* __restrict__ src, _Float16* __restrict__ dst,
                         int Nrows, int C, int stride, int coff) {
  long t = (long)blockIdx.x * blockDim.x + threadIdx.x;
  if (t >= (long)Nrows * 128) return;
  int n = (int)(t >> 7);
  int j = (int)(t & 127);
  float v = (j < C) ? src[(size_t)n * C + j] : 0.f;
  dst[(size_t)n * stride + coff + j] = (_Float16)v;
}

// ---------------- degree / norm precompute ---------------------------------
__global__ void k_deginit(float* __restrict__ dAll, float* __restrict__ dPre, int N) {
  int n = blockIdx.x * blockDim.x + threadIdx.x;
  if (n >= N) return;
  dAll[n] = 1.0f;   // GCNConv adds self loop of weight 1
  dPre[n] = 1.0f;
}

__global__ void k_degedge(const int* __restrict__ src, const int* __restrict__ dst,
                          float* __restrict__ dAll, float* __restrict__ dPre, int E) {
  int e = blockIdx.x * blockDim.x + threadIdx.x;
  if (e >= E) return;
  int s = src[e], d = dst[e];
  atomAddF(&dAll[d], 1.0f);
  if (s != d) atomAddF(&dPre[d], 1.0f);
}

__global__ void k_inv(const float* __restrict__ dAll, const float* __restrict__ dPre,
                      float* __restrict__ iAll, float* __restrict__ iPre, int N) {
  int n = blockIdx.x * blockDim.x + threadIdx.x;
  if (n >= N) return;
  iAll[n] = 1.0f / dAll[n];
  iPre[n] = 1.0f / dPre[n];
}

__global__ void k_norm(const int* __restrict__ src, const int* __restrict__ dst,
                       const float* __restrict__ dAll, const float* __restrict__ dPre,
                       float* __restrict__ nAll, float* __restrict__ nPre, int E) {
  int e = blockIdx.x * blockDim.x + threadIdx.x;
  if (e >= E) return;
  int s = src[e], d = dst[e];
  nAll[e] = rsqrtf(dAll[s] * dAll[d]);
  nPre[e] = (s != d) ? rsqrtf(dPre[s] * dPre[d]) : 0.0f;
}

// ---------------- WMMA GEMM -------------------------------------------------
// B (packed weights, NT*16 x KS) is staged in LDS once per block and shared by
// all 8 waves (8 M-tiles). LDS row stride is KS+8 halves to avoid bank
// conflicts for ds_load_b128 (row step = 272B = 4-bank rotation).
// One wave computes a 16 x (NT*16) row-stripe; epilogue writes xw and
// xw*inv[row] (self-loop init for the aggregation buffer).
template <int NT, int KS>
__global__ void k_gemm(const _Float16* __restrict__ A, const _Float16* __restrict__ B,
                       float* __restrict__ Cxw, float* __restrict__ Cagg,
                       const float* __restrict__ inv,
                       int Nrows, int Nout, int ldc) {
  constexpr int KT  = KS / 32;        // k tiles
  constexpr int LDB = KS + 8;         // padded LDS row stride (halves)
  constexpr int ROWS = NT * 16;
  __shared__ _Float16 sB[ROWS * LDB];

  // ---- stage B into LDS (all threads, before any divergence) ----
  {
    constexpr int CPR = KS / 8;       // 16-byte chunks per row
    for (int i = threadIdx.x; i < ROWS * CPR; i += blockDim.x) {
      int row = i / CPR;
      int col = i - row * CPR;
      *(uint4*)(sB + row * LDB + col * 8) =
          *(const uint4*)(B + (size_t)row * KS + col * 8);
    }
  }
  __syncthreads();

  int wv   = blockIdx.x * (blockDim.x >> 5) + (threadIdx.x >> 5);
  int lane = threadIdx.x & 31;
  int mT = (Nrows + 15) >> 4;
  if (wv >= mT) return;                    // wave-uniform: EXEC stays full
  int m0 = wv << 4;
  int laneHi = lane >> 4, laneLo = lane & 15;

  int arow = m0 + laneLo;
  if (arow >= Nrows) arow = Nrows - 1;     // clamp (results masked at store)

  const _Float16* arp = A + (size_t)arow * KS + laneHi * 16;
  // B fragment for tile nt, k-tile kt: row (nt*16 + laneLo), 16 halves at
  // offset kt*32 + laneHi*16
  const _Float16* brp = sB + laneLo * LDB + laneHi * 16;

  v8f acc[NT];
#pragma unroll
  for (int nt = 0; nt < NT; ++nt) acc[nt] = v8f{};

#pragma unroll
  for (int kt = 0; kt < KT; ++kt) {
    // A fragment in this code's k order (B matches): lane<16 holds K[0..15]; lane>=16 K[16..31]
    h8 alo = *(const h8*)(arp);
    h8 ahi = *(const h8*)(arp + 8);
    v16h av;
#pragma unroll
    for (int i = 0; i < 8; ++i) { av[i] = alo[i]; av[i + 8] = ahi[i]; }
#pragma unroll
    for (int nt = 0; nt < NT; ++nt) {
      v16h bv = *(const v16h*)(brp + nt * (16 * LDB) + kt * 32);
      acc[nt] = __builtin_amdgcn_wmma_f32_16x16x32_f16(false, av, false, bv,
                                                       (short)0, acc[nt], false, false);
    }
    arp += 32;
  }

  // C 16x16 f32: VGPR v, lanes0-15 -> M=m0+v, lanes16-31 -> M=m0+8+v
  int r0 = m0 + laneHi * 8;
#pragma unroll
  for (int v = 0; v < 8; ++v) {
    int r = r0 + v;
    if (r < Nrows) {
      float iv = inv[r];
      float* xwRow  = Cxw  + (size_t)r * ldc;
      float* aggRow = Cagg + (size_t)r * ldc;
#pragma unroll
      for (int nt = 0; nt < NT; ++nt) {
        int ccol = nt * 16 + laneLo;
        if (ccol < Nout) {
          float val = acc[nt][v];
          xwRow[ccol]  = val;
          aggRow[ccol] = val * iv;
        }
      }
    }
  }
}

// ---------------- aggregation (edge scatter) -------------------------------
__global__ void k_scatter(const float* __restrict__ xw, const int* __restrict__ src,
                          const int* __restrict__ dst, const float* __restrict__ nrm,
                          float* __restrict__ out, int E, int C) {
  long t = (long)blockIdx.x * blockDim.x + threadIdx.x;
  int e = (int)(t >> 5);
  int lane = (int)(t & 31);
  if (e >= E) return;
  float w = nrm[e];
  if (w == 0.f) return;
  int s = src[e], d = dst[e];
  const float* xs = xw + (size_t)s * C;
  float*       od = out + (size_t)d * C;
  for (int c = lane; c < C; c += 32) atomAddF(od + c, xs[c] * w);
}

// ---------------- epilogues ------------------------------------------------
__global__ void k_impute(const float* __restrict__ x, const float* __restrict__ agg,
                         const float* __restrict__ b, float* __restrict__ out,
                         int N, int C) {
  long t = (long)blockIdx.x * blockDim.x + threadIdx.x;
  if (t >= (long)N * C) return;
  int c = (int)(t % C);
  float xv = x[t];
  out[t] = (xv != 0.f) ? xv : (agg[t] + b[c]);
}

__global__ void k_biasrelu(const float* __restrict__ agg, const float* __restrict__ b,
                           float* __restrict__ out, int N, int C) {
  long t = (long)blockIdx.x * blockDim.x + threadIdx.x;
  if (t >= (long)N * C) return;
  int c = (int)(t % C);
  float v = agg[t] + b[c];
  out[t] = v > 0.f ? v : 0.f;
}

__global__ void k_skiprelu(const float* __restrict__ agg, const float* __restrict__ b,
                           float* __restrict__ inout, int N, int C) {
  long t = (long)blockIdx.x * blockDim.x + threadIdx.x;
  if (t >= (long)N * C) return;
  int c = (int)(t % C);
  float v = agg[t] + b[c] + inout[t];
  inout[t] = v > 0.f ? v : 0.f;
}

__global__ void k_final(const float* __restrict__ agg, const float* __restrict__ b,
                        const float* __restrict__ mask, float* __restrict__ out, int N) {
  long t = (long)blockIdx.x * blockDim.x + threadIdx.x;
  if (t >= (long)N * 48) return;
  int n = (int)(t / 48);
  int c = (int)(t % 48);
  float v = agg[t] + b[c];
  float sg = 1.0f / (1.0f + __expf(-v));
  out[t] = sg * 255.0f * mask[n];
}

// ---------------- host orchestration ---------------------------------------
extern "C" void kernel_launch(void* const* d_in, const int* in_sizes, int n_in,
                              void* d_out, int out_size, void* d_ws, size_t ws_size,
                              hipStream_t stream) {
  const float* x    = (const float*)d_in[0];
  const int*   ei   = (const int*)d_in[1];
  const float* mask = (const float*)d_in[3];
  const float* preW = (const float*)d_in[4];
  const float* preB = (const float*)d_in[5];
  const float* W1s  = (const float*)d_in[6];
  const float* b1s  = (const float*)d_in[7];
  const float* W2s  = (const float*)d_in[8];
  const float* b2s  = (const float*)d_in[9];
  const float* Wf   = (const float*)d_in[10];
  const float* bf   = (const float*)d_in[11];

  const int N = in_sizes[3];          // 215820 (mask)
  const int E = in_sizes[1] / 2;      // 1726560
  const int C = 100;
  const int depth = 5;
  const int* srcI = ei;
  const int* dstI = ei + E;

  char* wsp = (char*)d_ws;
  size_t off = 0;
  auto alloc = [&](size_t bytes) -> void* {
    void* p = wsp + off;
    off += (bytes + 255) & ~(size_t)255;
    return p;
  };

  _Float16* wtPre = (_Float16*)alloc((size_t)112 * 128 * 2);
  _Float16* wt1[5];
  _Float16* wt2[5];
  for (int i = 0; i < depth; ++i) wt1[i] = (_Float16*)alloc((size_t)112 * 128 * 2);
  for (int i = 0; i < depth; ++i) wt2[i] = (_Float16*)alloc((size_t)112 * 128 * 2);
  _Float16* wtF = (_Float16*)alloc((size_t)48 * 256 * 2);

  float* degAll = (float*)alloc((size_t)N * 4);
  float* degPre = (float*)alloc((size_t)N * 4);
  float* invAll = (float*)alloc((size_t)N * 4);
  float* invPre = (float*)alloc((size_t)N * 4);
  float* nrmAll = (float*)alloc((size_t)E * 4);
  float* nrmPre = (float*)alloc((size_t)E * 4);
  float* xcur   = (float*)alloc((size_t)N * C * 4);
  float* buf1   = (float*)alloc((size_t)N * C * 4);
  float* buf2   = (float*)alloc((size_t)N * C * 4);
  _Float16* A16 = (_Float16*)alloc((size_t)N * 256 * 2);

  const int TB = 256;
  auto nb = [](long work, int tb) { return (unsigned)((work + tb - 1) / tb); };

  // --- weight packing ---
  k_pack<<<nb(112 * 128, TB), TB, 0, stream>>>(preW, wtPre, 100, 100, 100, 112, 128, 0, 0);
  for (int i = 0; i < depth; ++i) {
    k_pack<<<nb(112 * 128, TB), TB, 0, stream>>>(W1s + (size_t)i * C * C, wt1[i], 100, 100, 100, 112, 128, 0, 0);
    k_pack<<<nb(112 * 128, TB), TB, 0, stream>>>(W2s + (size_t)i * C * C, wt2[i], 100, 100, 100, 112, 128, 0, 0);
  }
  k_pack<<<nb(48 * 128, TB), TB, 0, stream>>>(Wf, wtF, 100, 48, 48, 48, 256, 0, 0);
  k_pack<<<nb(48 * 128, TB), TB, 0, stream>>>(Wf, wtF, 100, 48, 48, 48, 256, 128, 100);

  // --- degrees & norms ---
  k_deginit<<<nb(N, TB), TB, 0, stream>>>(degAll, degPre, N);
  k_degedge<<<nb(E, TB), TB, 0, stream>>>(srcI, dstI, degAll, degPre, E);
  k_inv<<<nb(N, TB), TB, 0, stream>>>(degAll, degPre, invAll, invPre, N);
  k_norm<<<nb(E, TB), TB, 0, stream>>>(srcI, dstI, degAll, degPre, nrmAll, nrmPre, E);

  int mT = (N + 15) / 16;
  // GEMM writes xw into buf1 and xw*inv into buf2 (self-loop init), then the
  // edge scatter accumulates into buf2.
  auto gemm100 = [&](const _Float16* Bp, const float* inv) {
    k_gemm<7, 128><<<nb(mT, 8), TB, 0, stream>>>(A16, Bp, buf1, buf2, inv, N, 100, 100);
  };
  auto conv16 = [&](const float* s, int stride, int coff) {
    k_conv16<<<nb((long)N * 128, TB), TB, 0, stream>>>(s, A16, N, C, stride, coff);
  };
  auto scatter = [&](const float* nrm, int Cc) {
    k_scatter<<<nb((long)E * 32, TB), TB, 0, stream>>>(buf1, srcI, dstI, nrm, buf2, E, Cc);
  };

  // --- pre conv (self loops masked) + imputation ---
  conv16(x, 128, 0);
  gemm100(wtPre, invPre);
  scatter(nrmPre, 100);
  k_impute<<<nb((long)N * C, TB), TB, 0, stream>>>(x, buf2, preB, xcur, N, C);

  // --- residual blocks ---
  for (int i = 0; i < depth; ++i) {
    conv16(xcur, 128, 0);
    gemm100(wt1[i], invAll);
    scatter(nrmAll, 100);
    k_biasrelu<<<nb((long)N * C, TB), TB, 0, stream>>>(buf2, b1s + (size_t)i * C, buf1, N, C);  // h -> buf1
    conv16(buf1, 128, 0);
    gemm100(wt2[i], invAll);
    scatter(nrmAll, 100);
    k_skiprelu<<<nb((long)N * C, TB), TB, 0, stream>>>(buf2, b2s + (size_t)i * C, xcur, N, C);
  }

  // --- final conv on concat [xcur | x0] via K=256 GEMM ---
  conv16(xcur, 256, 0);
  conv16(x, 256, 128);
  k_gemm<3, 256><<<nb(mT, 8), TB, 0, stream>>>(A16, wtF, buf1, buf2, invAll, N, 48, 48);
  scatter(nrmAll, 48);
  k_final<<<nb((long)N * 48, TB), TB, 0, stream>>>(buf2, bf, mask, (float*)d_out, N);
}
